// MixedAttentionLayer_46024869544253
// MI455X (gfx1250) — matrix operand order repaired and checked by reading, not compiled, not run on hardware
//
#include <hip/hip_runtime.h>

typedef _Float16 f16;
typedef __attribute__((ext_vector_type(16))) _Float16 v16h;
typedef __attribute__((ext_vector_type(8)))  float    v8f;

constexpr int   D_     = 256;
constexpr int   MC     = 50;
constexpr int   NPATH  = 65536;
constexpr int   NTOK   = NPATH + MC;   // 65586
constexpr float SCALE  = 0.0625f;      // 256^-0.5

static __device__ __forceinline__ float wred(float v) {
    for (int off = 16; off; off >>= 1) v += __shfl_xor(v, off, 32);
    return v;
}

// ---------------------------------------------------------------------------
// Generic WMMA GEMM:  C(f32) = A(f16) @ B(f16) [+ epilogue]
// Block tile 128x128, K-step 32; 8 waves arranged 4(M) x 2(N); each wave owns
// a 32x64 region = 2x4 fragments of v_wmma_f32_16x16x32_f16.
// ---------------------------------------------------------------------------
constexpr int TM = 128, TN = 128, TK = 32;
constexpr int AST = TK + 2, BST = TK + 2;   // LDS padding (bank spread)

enum { EPI_NONE = 0, EPI_BIAS, EPI_BIAS_TANH, EPI_BIAS_SIGMUL, EPI_ROWSCALE_ADDAUX, EPI_ATOMIC };

template <int EPI, bool ATRANS>
__global__ __launch_bounds__(256) void gemm_f16(
    const f16* __restrict__ A, const f16* __restrict__ B, float* __restrict__ C,
    int Mdim, int Ndim, int Kdim, int lda, int ldb, int ldc,
    const float* __restrict__ bias, const float* __restrict__ aux,
    const float* __restrict__ rowscale, int kChunk)
{
    __shared__ f16 As[TM * AST];
    __shared__ f16 Bs[TN * BST];
    const int tid  = threadIdx.x;
    const int lane = tid & 31;
    const int wave = tid >> 5;
    const int half = lane >> 4;
    const int l16  = lane & 15;
    const int wm   = wave & 3;    // 0..3  -> M offset wm*32
    const int wn   = wave >> 2;   // 0..1  -> N offset wn*64
    const int bM0  = blockIdx.y * TM;
    const int bN0  = blockIdx.x * TN;

    int k0 = 0, kEnd = Kdim;
    if (kChunk > 0) {
        k0 = blockIdx.z * kChunk;
        kEnd = k0 + kChunk; if (kEnd > Kdim) kEnd = Kdim;
    }

    v8f acc[2][4];
    for (int r = 0; r < 2; r++)
        for (int c = 0; c < 4; c++)
            for (int v = 0; v < 8; v++) acc[r][c][v] = 0.f;

    for (int kt = k0; kt < kEnd; kt += TK) {
        // ---- stage A tile (TM x TK) ----
        if (!ATRANS) {
            #pragma unroll
            for (int i = 0; i < 8; i++) {
                int p  = tid + i * 256;         // pair index, 16 pairs per row
                int m  = p >> 4;
                int kk = (p & 15) << 1;
                int gm = bM0 + m;
                f16 v0 = (f16)0.f, v1 = (f16)0.f;
                if (gm < Mdim) {
                    const f16* s = A + (size_t)gm * lda + (kt + kk);
                    v0 = s[0]; v1 = s[1];
                }
                As[m * AST + kk]     = v0;
                As[m * AST + kk + 1] = v1;
            }
        } else {  // A[m,k] = Asrc[k*lda + m]  (column-major view, coalesced in m)
            #pragma unroll
            for (int i = 0; i < 16; i++) {
                int e  = tid + i * 256;
                int m  = e & 127;
                int kk = e >> 7;
                int gm = bM0 + m;
                f16 v = (f16)0.f;
                if (gm < Mdim) v = A[(size_t)(kt + kk) * lda + gm];
                As[m * AST + kk] = v;
            }
        }
        // ---- stage B tile transposed: Bs[n][k] ----
        #pragma unroll
        for (int i = 0; i < 16; i++) {
            int e  = tid + i * 256;
            int n  = e & 127;
            int kk = e >> 7;
            int gn = bN0 + n;
            f16 v = (f16)0.f;
            if (gn < Ndim) v = B[(size_t)(kt + kk) * ldb + gn];
            Bs[n * BST + kk] = v;
        }
        __syncthreads();

        // ---- fragment loads per ISA 7.12.2 layouts ----
        union U16 { v16h v; unsigned u[8]; };
        U16 af[2], bfr[4];
        const unsigned* Asu = (const unsigned*)As;
        const unsigned* Bsu = (const unsigned*)Bs;
        #pragma unroll
        for (int r = 0; r < 2; r++) {
            int mrow  = wm * 32 + r * 16 + l16;
            int base0 = (mrow * AST + half * 8) >> 1;        // K = half*8 + 0..7
            int base1 = (mrow * AST + 16 + half * 8) >> 1;   // K = 16 + half*8 + 0..7
            #pragma unroll
            for (int t = 0; t < 4; t++) { af[r].u[t] = Asu[base0 + t]; af[r].u[4 + t] = Asu[base1 + t]; }
        }
        #pragma unroll
        for (int c = 0; c < 4; c++) {
            int ncol = wn * 64 + c * 16 + l16;
            int base0 = (ncol * BST + half * 8) >> 1;        // K = half*8 + 0..7
            int base1 = (ncol * BST + 16 + half * 8) >> 1;   // K = 16 + half*8 + 0..7
            #pragma unroll
            for (int t = 0; t < 4; t++) { bfr[c].u[t] = Bsu[base0 + t]; bfr[c].u[4 + t] = Bsu[base1 + t]; }
        }
        #pragma unroll
        for (int r = 0; r < 2; r++)
            #pragma unroll
            for (int c = 0; c < 4; c++)
                acc[r][c] = __builtin_amdgcn_wmma_f32_16x16x32_f16(
                    false, af[r].v, false, bfr[c].v, (short)0, acc[r][c], false, false);
        __syncthreads();
    }

    // ---- epilogue: C element (row = base + half*8 + v, col = base + l16) ----
    #pragma unroll
    for (int r = 0; r < 2; r++)
        #pragma unroll
        for (int c = 0; c < 4; c++) {
            int col = bN0 + wn * 64 + c * 16 + l16;
            if (col >= Ndim) continue;
            #pragma unroll
            for (int v = 0; v < 8; v++) {
                int row = bM0 + wm * 32 + r * 16 + half * 8 + v;
                if (row >= Mdim) continue;
                size_t idx = (size_t)row * ldc + col;
                float val = acc[r][c][v];
                if constexpr (EPI == EPI_BIAS)           C[idx] = val + bias[col];
                else if constexpr (EPI == EPI_BIAS_TANH) C[idx] = tanhf(val + bias[col]);
                else if constexpr (EPI == EPI_BIAS_SIGMUL)
                    C[idx] = (1.f / (1.f + expf(-(val + bias[col])))) * aux[idx];
                else if constexpr (EPI == EPI_ROWSCALE_ADDAUX)
                    C[idx] = val * rowscale[row] + aux[idx];
                else if constexpr (EPI == EPI_ATOMIC)    atomicAdd(&C[idx], val);
                else                                     C[idx] = val;
            }
        }
}

// ---------------------------------------------------------------------------
// Utility kernels
// ---------------------------------------------------------------------------
__global__ __launch_bounds__(256) void cvt_f16_kernel(const float* __restrict__ src,
    f16* __restrict__ dst, long long total, int cols, int srcLd)
{
    for (long long i = blockIdx.x * 256LL + threadIdx.x; i < total;
         i += (long long)gridDim.x * 256LL) {
        long long r = i / cols;
        int c = (int)(i - r * cols);
        dst[i] = (f16)src[r * (long long)srcLd + c];
    }
}

__global__ __launch_bounds__(256) void zero_kernel(float* __restrict__ p, long long n)
{
    for (long long i = blockIdx.x * 256LL + threadIdx.x; i < n;
         i += (long long)gridDim.x * 256LL) p[i] = 0.f;
}

__global__ __launch_bounds__(256) void softplus_kernel(const float* __restrict__ x,
                                                       float* __restrict__ sp)
{
    int d = threadIdx.x;
    float v = x[d];
    sp[d] = (v > 20.f) ? v : log1pf(expf(v));
}

// ---------------------------------------------------------------------------
// Attention kernels
// ---------------------------------------------------------------------------
// 50x50 self attention over cell tokens -> osc (50x256)
__global__ __launch_bounds__(256) void self_cell_attn(const float* __restrict__ qkv,
                                                      float* __restrict__ osc)
{
    __shared__ float S[MC][MC + 2];
    int tid = threadIdx.x;
    for (int idx = tid; idx < MC * MC; idx += 256) {
        int i = idx / MC, j = idx % MC;
        const float* q = qkv + (size_t)i * 768;
        const float* k = qkv + (size_t)j * 768 + 256;
        float d = 0.f;
        for (int c = 0; c < D_; c++) d += q[c] * k[c];
        S[i][j] = d * SCALE;
    }
    __syncthreads();
    if (tid < MC) {
        float mx = -3.4e38f;
        for (int j = 0; j < MC; j++) mx = fmaxf(mx, S[tid][j]);
        float se = 0.f;
        for (int j = 0; j < MC; j++) { S[tid][j] = expf(S[tid][j] - mx); se += S[tid][j]; }
        float inv = 1.f / se;
        for (int j = 0; j < MC; j++) S[tid][j] *= inv;
    }
    __syncthreads();
    for (int idx = tid; idx < MC * D_; idx += 256) {
        int i = idx >> 8, d = idx & 255;
        float acc = 0.f;
        for (int j = 0; j < MC; j++) acc += S[i][j] * qkv[(size_t)j * 768 + 512 + d];
        osc[idx] = acc;
    }
}

// 65536 path queries x 50 cell keys -> ocp (65536x256).  One wave per query.
__global__ __launch_bounds__(256) void cross_path_attn(const float* __restrict__ qkv,
                                                       float* __restrict__ ocp)
{
    int wave = threadIdx.x >> 5, lane = threadIdx.x & 31;
    int p = blockIdx.x * 8 + wave;
    const float* qrow = qkv + (size_t)(MC + p) * 768;
    float q[8];
    #pragma unroll
    for (int t = 0; t < 8; t++) q[t] = qrow[lane * 8 + t];
    float sc[MC];
    for (int j = 0; j < MC; j++) {
        const float* krow = qkv + (size_t)j * 768 + 256;
        float d = 0.f;
        #pragma unroll
        for (int t = 0; t < 8; t++) d += q[t] * krow[lane * 8 + t];
        sc[j] = wred(d) * SCALE;
    }
    float mx = -3.4e38f;
    for (int j = 0; j < MC; j++) mx = fmaxf(mx, sc[j]);
    float se = 0.f;
    for (int j = 0; j < MC; j++) { sc[j] = expf(sc[j] - mx); se += sc[j]; }
    float inv = 1.f / se;
    float acc[8] = {0, 0, 0, 0, 0, 0, 0, 0};
    for (int j = 0; j < MC; j++) {
        const float* vrow = qkv + (size_t)j * 768 + 512;
        float w = sc[j] * inv;
        #pragma unroll
        for (int t = 0; t < 8; t++) acc[t] += w * vrow[lane * 8 + t];
    }
    float* orow = ocp + (size_t)p * D_;
    #pragma unroll
    for (int t = 0; t < 8; t++) orow[lane * 8 + t] = acc[t];
}

// cac logits: S[i][j] = scale * qc_i . kp_j  (50 x 65536), kp staged in LDS
__global__ __launch_bounds__(256) void cac_scores(const float* __restrict__ qkv,
                                                  float* __restrict__ S)
{
    __shared__ float kpS[32 * 256];
    int j0 = blockIdx.x * 32;
    for (int e = threadIdx.x; e < 32 * 256; e += 256) {
        int jj = e >> 8, d = e & 255;
        kpS[e] = qkv[(size_t)(MC + j0 + jj) * 768 + 256 + d];
    }
    __syncthreads();
    for (int idx = threadIdx.x; idx < MC * 32; idx += 256) {
        int i = idx % MC, jj = idx / MC;
        const float* q = qkv + (size_t)i * 768;
        float acc = 0.f;
        for (int c = 0; c < D_; c++) acc += q[c] * kpS[jj * 256 + c];
        S[(size_t)i * NPATH + j0 + jj] = acc * SCALE;
    }
}

__global__ __launch_bounds__(256) void cac_rowstats(const float* __restrict__ S,
                                                    float* __restrict__ rmax,
                                                    float* __restrict__ rsum)
{
    int i = blockIdx.x;
    __shared__ float red[256];
    const float* row = S + (size_t)i * NPATH;
    float mx = -3.4e38f;
    for (int j = threadIdx.x; j < NPATH; j += 256) mx = fmaxf(mx, row[j]);
    red[threadIdx.x] = mx; __syncthreads();
    for (int s = 128; s; s >>= 1) {
        if (threadIdx.x < s) red[threadIdx.x] = fmaxf(red[threadIdx.x], red[threadIdx.x + s]);
        __syncthreads();
    }
    mx = red[0]; __syncthreads();
    float se = 0.f;
    for (int j = threadIdx.x; j < NPATH; j += 256) se += expf(row[j] - mx);
    red[threadIdx.x] = se; __syncthreads();
    for (int s = 128; s; s >>= 1) {
        if (threadIdx.x < s) red[threadIdx.x] += red[threadIdx.x + s];
        __syncthreads();
    }
    if (threadIdx.x == 0) { rmax[i] = mx; rsum[i] = red[0]; }
}

__global__ __launch_bounds__(256) void cac_softmax_f16(const float* __restrict__ S,
    const float* __restrict__ rmax, const float* __restrict__ rsum, f16* __restrict__ P)
{
    long long n = (long long)MC * NPATH;
    for (long long idx = blockIdx.x * 256LL + threadIdx.x; idx < n;
         idx += (long long)gridDim.x * 256LL) {
        int i = (int)(idx >> 16);  // NPATH == 65536
        P[idx] = (f16)(expf(S[idx] - rmax[i]) / rsum[i]);
    }
}

// ---------------------------------------------------------------------------
// Linear attention feature map: l = (relu(v)+1e-6)/sp; out = l^3/||l^3|| * ||l||
// One wave per row.
// ---------------------------------------------------------------------------
__global__ __launch_bounds__(256) void linattn_feat(const float* __restrict__ src,
    const float* __restrict__ sp, f16* __restrict__ dst)
{
    int wave = threadIdx.x >> 5, lane = threadIdx.x & 31;
    int p = blockIdx.x * 8 + wave;
    const float* row = src + (size_t)p * 768;
    float l[8];
    #pragma unroll
    for (int t = 0; t < 8; t++) {
        float v = row[lane * 8 + t];
        v = fmaxf(v, 0.f) + 1e-6f;
        l[t] = v / sp[lane * 8 + t];
    }
    float n1 = 0.f;
    #pragma unroll
    for (int t = 0; t < 8; t++) n1 += l[t] * l[t];
    n1 = sqrtf(wred(n1));
    float c3[8]; float n3 = 0.f;
    #pragma unroll
    for (int t = 0; t < 8; t++) { c3[t] = l[t] * l[t] * l[t]; n3 += c3[t] * c3[t]; }
    n3 = sqrtf(wred(n3));
    float s = n1 / n3;
    f16* drow = dst + (size_t)p * D_;
    #pragma unroll
    for (int t = 0; t < 8; t++) drow[lane * 8 + t] = (f16)(c3[t] * s);
}

__global__ __launch_bounds__(256) void ksum_kernel(const f16* __restrict__ kl,
                                                   float* __restrict__ ksum)
{
    int d = threadIdx.x;
    int r0 = blockIdx.x * 256;
    float acc = 0.f;
    for (int r = r0; r < r0 + 256; r++) acc += (float)kl[(size_t)r * D_ + d];
    atomicAdd(&ksum[d], acc);
}

__global__ __launch_bounds__(256) void zfac_kernel(const f16* __restrict__ ql,
    const float* __restrict__ ksum, float* __restrict__ zfac)
{
    int wave = threadIdx.x >> 5, lane = threadIdx.x & 31;
    int p = blockIdx.x * 8 + wave;
    const f16* row = ql + (size_t)p * D_;
    float acc = 0.f;
    #pragma unroll
    for (int t = 0; t < 8; t++) acc += (float)row[lane * 8 + t] * ksum[lane * 8 + t];
    acc = wred(acc);
    if (lane == 0) zfac[p] = 1.f / (acc + 1e-6f);
}

// ---------------------------------------------------------------------------
// Depthwise 5x5 conv (SAME) on vp viewed as 256x256 spatial x 256 channels
// ---------------------------------------------------------------------------
__global__ __launch_bounds__(256) void dwconv_kernel(const float* __restrict__ qkv,
    const float* __restrict__ w, const float* __restrict__ b, float* __restrict__ fm)
{
    int idx = blockIdx.x * 256 + threadIdx.x;
    int c = idx & 255;
    int p = idx >> 8;
    int y = p >> 8, x = p & 255;
    float acc = b[c];
    #pragma unroll
    for (int dy = 0; dy < 5; dy++) {
        int yy = y + dy - 2; if (yy < 0 || yy > 255) continue;
        #pragma unroll
        for (int dx = 0; dx < 5; dx++) {
            int xx = x + dx - 2; if (xx < 0 || xx > 255) continue;
            acc += w[c * 25 + dy * 5 + dx] *
                   qkv[(size_t)(MC + yy * 256 + xx) * 768 + 512 + c];
        }
    }
    fm[idx] = acc;
}

// ---------------------------------------------------------------------------
// Gating score reduce: A[r] = gate[r,:] . Wc + bc   (one wave per row)
// ---------------------------------------------------------------------------
__global__ __launch_bounds__(256) void wc_reduce(const float* __restrict__ gate,
    const float* __restrict__ Wc, const float* __restrict__ bc, float* __restrict__ A)
{
    int wave = threadIdx.x >> 5, lane = threadIdx.x & 31;
    int r = blockIdx.x * 8 + wave;
    const float* g = gate + (size_t)r * D_;
    float acc = 0.f;
    #pragma unroll
    for (int t = 0; t < 8; t++) acc += g[lane * 8 + t] * Wc[lane * 8 + t];
    acc = wred(acc);
    if (lane == 0) A[r] = acc + bc[0];
}

// Small (50-row) gated scoring, all scalar (6.6M MAC total)
__global__ __launch_bounds__(256) void gated_small(const float* __restrict__ t,
    const float* __restrict__ Wa, const float* __restrict__ ba,
    const float* __restrict__ Wb, const float* __restrict__ bb,
    const float* __restrict__ Wc, const float* __restrict__ bc, float* __restrict__ A)
{
    __shared__ float Ash[MC];
    int tid = threadIdx.x;
    if (tid < MC) Ash[tid] = 0.f;
    __syncthreads();
    for (int idx = tid; idx < MC * D_; idx += 256) {
        int i = idx >> 8, d = idx & 255;
        const float* tr = t + (size_t)i * D_;
        float s1 = ba[d], s2 = bb[d];
        for (int c = 0; c < D_; c++) {
            float tv = tr[c];
            s1 += tv * Wa[c * D_ + d];
            s2 += tv * Wb[c * D_ + d];
        }
        float val = tanhf(s1) * (1.f / (1.f + expf(-s2))) * Wc[d];
        atomicAdd(&Ash[i], val);
    }
    __syncthreads();
    if (tid < MC) A[tid] = Ash[tid] + bc[0];
}

// ---------------------------------------------------------------------------
// Softmax pooling: h = softmax(A)^T @ t
// ---------------------------------------------------------------------------
__global__ __launch_bounds__(256) void pool_stats(const float* __restrict__ A, int n,
                                                  float* __restrict__ stats)
{
    __shared__ float red[256];
    float mx = -3.4e38f;
    for (int i = threadIdx.x; i < n; i += 256) mx = fmaxf(mx, A[i]);
    red[threadIdx.x] = mx; __syncthreads();
    for (int s = 128; s; s >>= 1) {
        if (threadIdx.x < s) red[threadIdx.x] = fmaxf(red[threadIdx.x], red[threadIdx.x + s]);
        __syncthreads();
    }
    mx = red[0]; __syncthreads();
    float se = 0.f;
    for (int i = threadIdx.x; i < n; i += 256) se += expf(A[i] - mx);
    red[threadIdx.x] = se; __syncthreads();
    for (int s = 128; s; s >>= 1) {
        if (threadIdx.x < s) red[threadIdx.x] += red[threadIdx.x + s];
        __syncthreads();
    }
    if (threadIdx.x == 0) { stats[0] = mx; stats[1] = red[0]; }
}

__global__ __launch_bounds__(256) void pool_apply(const float* __restrict__ A,
    const float* __restrict__ t, const float* __restrict__ stats,
    float* __restrict__ h, int n)
{
    int d = threadIdx.x;
    int r0 = blockIdx.x * 256;
    int r1 = r0 + 256; if (r1 > n) r1 = n;
    float mx = stats[0];
    float inv = 1.f / stats[1];
    float acc = 0.f;
    for (int r = r0; r < r1; r++) acc += expf(A[r] - mx) * t[(size_t)r * D_ + d];
    atomicAdd(&h[d], acc * inv);
}

// ---------------------------------------------------------------------------
// Final fuse MLPs (tiny): out[0..256)=fusion_cross, out[256..512)=fusion_self
// ---------------------------------------------------------------------------
__global__ __launch_bounds__(256) void final_mlp(
    const float* __restrict__ hcc, const float* __restrict__ hcp,
    const float* __restrict__ hsc, const float* __restrict__ hsp,
    const float* __restrict__ W1, const float* __restrict__ b1,
    const float* __restrict__ W2, const float* __restrict__ b2,
    const float* __restrict__ W3a, const float* __restrict__ b3a,
    const float* __restrict__ W3b, const float* __restrict__ b3b,
    float* __restrict__ out)
{
    __shared__ float f[512], g[256];
    int d = threadIdx.x;
    for (int pass = 0; pass < 2; pass++) {
        const float* hA = pass ? hsc : hcc;   // h1 stream (W1)
        const float* hB = pass ? hsp : hcp;   // h2 stream (W2)
        float s = b1[d];
        for (int c = 0; c < 256; c++) s += hA[c] * W1[c * 256 + d];
        f[d] = fmaxf(s, 0.f);
        s = b2[d];
        for (int c = 0; c < 256; c++) s += hB[c] * W2[c * 256 + d];
        f[256 + d] = fmaxf(s, 0.f);
        __syncthreads();
        s = b3a[d];
        for (int c = 0; c < 512; c++) s += f[c] * W3a[c * 256 + d];
        g[d] = fmaxf(s, 0.f);
        __syncthreads();
        s = b3b[d];
        for (int c = 0; c < 256; c++) s += g[c] * W3b[c * 256 + d];
        out[pass * 256 + d] = fmaxf(s, 0.f);
        __syncthreads();
    }
}

// ---------------------------------------------------------------------------
// Host orchestration
// ---------------------------------------------------------------------------
extern "C" void kernel_launch(void* const* d_in, const int* in_sizes, int n_in,
                              void* d_out, int out_size, void* d_ws, size_t ws_size,
                              hipStream_t stream)
{
    (void)in_sizes; (void)n_in; (void)out_size; (void)ws_size;
    const float* x      = (const float*)d_in[0];
    const float* Wqkv   = (const float*)d_in[1];
    const float* scl    = (const float*)d_in[2];
    const float* dwc_w  = (const float*)d_in[3];
    const float* dwc_b  = (const float*)d_in[4];
    const float* Wa     = (const float*)d_in[5];
    const float* ba     = (const float*)d_in[6];
    const float* Wb     = (const float*)d_in[7];
    const float* bb     = (const float*)d_in[8];
    const float* Wc     = (const float*)d_in[9];
    const float* bc     = (const float*)d_in[10];
    const float* W1     = (const float*)d_in[11];
    const float* b1     = (const float*)d_in[12];
    const float* W2     = (const float*)d_in[13];
    const float* b2     = (const float*)d_in[14];
    const float* W3a    = (const float*)d_in[15];
    const float* b3a    = (const float*)d_in[16];
    const float* W3b    = (const float*)d_in[17];
    const float* b3b    = (const float*)d_in[18];
    const float* Wf     = (const float*)d_in[19];
    const float* bfv    = (const float*)d_in[20];
    float* out = (float*)d_out;

    // ---- workspace carve-up ----
    uintptr_t base = (uintptr_t)d_ws;
    size_t off = 0;
    auto alloc = [&](size_t bytes) -> void* {
        void* p = (void*)(base + off);
        off = (off + bytes + 255) & ~(size_t)255;
        return p;
    };
    f16*   xbf    = (f16*)alloc((size_t)NTOK * D_ * 2);       // later reused as tmpbf
    f16*   wqkvbf = (f16*)alloc((size_t)256 * 768 * 2);
    f16*   wabf   = (f16*)alloc((size_t)65536 * 2);
    f16*   wbbf   = (f16*)alloc((size_t)65536 * 2);
    f16*   wfbf   = (f16*)alloc((size_t)65536 * 2);
    float* qkvf   = (float*)alloc((size_t)NTOK * 768 * 4);    // later reused as gate buffer
    f16*   qlbf   = (f16*)alloc((size_t)NPATH * D_ * 2);      // later reused as ocp_bf
    f16*   klbf   = (f16*)alloc((size_t)NPATH * D_ * 2);      // later reused as osp_bf
    f16*   vpbf   = (f16*)alloc((size_t)NPATH * D_ * 2);
    float* fmbuf  = (float*)alloc((size_t)NPATH * D_ * 4);    // fm -> tmp -> osp (in place)
    float* ocp    = (float*)alloc((size_t)NPATH * D_ * 4);
    float* kvf    = (float*)alloc((size_t)256 * 256 * 4);
    f16*   kvbf   = (f16*)alloc((size_t)256 * 256 * 2);
    float* Sbuf   = (float*)alloc((size_t)MC * NPATH * 4);
    f16*   Pbf    = (f16*)alloc((size_t)MC * NPATH * 2);
    float* spbuf  = (float*)alloc(256 * 4);
    float* ksum   = (float*)alloc(256 * 4);
    float* zfac   = (float*)alloc((size_t)NPATH * 4);
    float* rmax   = (float*)alloc(64 * 4);
    float* rsum   = (float*)alloc(64 * 4);
    float* Acp    = (float*)alloc((size_t)NPATH * 4);
    float* Asp    = (float*)alloc((size_t)NPATH * 4);
    float* Acc    = (float*)alloc(64 * 4);
    float* Asc    = (float*)alloc(64 * 4);
    float* occ    = (float*)alloc((size_t)MC * D_ * 4);
    float* osc    = (float*)alloc((size_t)MC * D_ * 4);
    float* hbuf   = (float*)alloc(1024 * 4);                  // hcc|hcp|hsc|hsp
    float* stats  = (float*)alloc(8 * 4);
    float* hcc = hbuf, *hcp = hbuf + 256, *hsc = hbuf + 512, *hsp = hbuf + 768;
    float* gate = qkvf;     // overlay: qkv f32 dead by gating time
    f16* tmpbf  = xbf;      // overlay: x bf16 dead after qkv GEMM
    f16* ocpbf  = qlbf;     // overlay: ql dead after out_s GEMM
    f16* ospbf  = klbf;     // overlay: kl dead after kv GEMM / ksum

    // ---- 1. conversions & softplus ----
    cvt_f16_kernel<<<4096, 256, 0, stream>>>(x, xbf, (long long)NTOK * D_, D_, D_);
    cvt_f16_kernel<<<512, 256, 0, stream>>>(Wqkv, wqkvbf, (long long)256 * 768, 768, 768);
    cvt_f16_kernel<<<256, 256, 0, stream>>>(Wa, wabf, 65536, 256, 256);
    cvt_f16_kernel<<<256, 256, 0, stream>>>(Wb, wbbf, 65536, 256, 256);
    cvt_f16_kernel<<<256, 256, 0, stream>>>(Wf, wfbf, 65536, 256, 256);
    softplus_kernel<<<1, 256, 0, stream>>>(scl, spbuf);

    // ---- 2. qkv = x @ W_qkv ----
    gemm_f16<EPI_NONE, false><<<dim3(6, 513, 1), 256, 0, stream>>>(
        xbf, wqkvbf, qkvf, NTOK, 768, 256, 256, 768, 768, nullptr, nullptr, nullptr, 0);
    cvt_f16_kernel<<<4096, 256, 0, stream>>>(qkvf + (size_t)MC * 768 + 512, vpbf,
                                             (long long)NPATH * D_, D_, 768);

    // ---- 3. small / cross attention ----
    self_cell_attn<<<1, 256, 0, stream>>>(qkvf, osc);
    cross_path_attn<<<NPATH / 8, 256, 0, stream>>>(qkvf, ocp);

    // ---- 4. cac: softmax(qc@kp^T) @ vp via split-K WMMA ----
    cac_scores<<<NPATH / 32, 256, 0, stream>>>(qkvf, Sbuf);
    cac_rowstats<<<MC, 256, 0, stream>>>(Sbuf, rmax, rsum);
    cac_softmax_f16<<<4096, 256, 0, stream>>>(Sbuf, rmax, rsum, Pbf);
    zero_kernel<<<64, 256, 0, stream>>>(occ, (long long)MC * D_);
    gemm_f16<EPI_ATOMIC, false><<<dim3(2, 1, 8), 256, 0, stream>>>(
        Pbf, vpbf, occ, MC, 256, NPATH, NPATH, 256, 256, nullptr, nullptr, nullptr, 8192);

    // ---- 5. linear attention ----
    linattn_feat<<<NPATH / 8, 256, 0, stream>>>(qkvf + (size_t)MC * 768, spbuf, qlbf);
    linattn_feat<<<NPATH / 8, 256, 0, stream>>>(qkvf + (size_t)MC * 768 + 256, spbuf, klbf);
    zero_kernel<<<1, 256, 0, stream>>>(ksum, 256);
    ksum_kernel<<<256, 256, 0, stream>>>(klbf, ksum);
    zfac_kernel<<<NPATH / 8, 256, 0, stream>>>(qlbf, ksum, zfac);
    zero_kernel<<<256, 256, 0, stream>>>(kvf, 65536);
    gemm_f16<EPI_ATOMIC, true><<<dim3(2, 2, 8), 256, 0, stream>>>(
        klbf, vpbf, kvf, 256, 256, NPATH, 256, 256, 256, nullptr, nullptr, nullptr, 8192);
    cvt_f16_kernel<<<64, 256, 0, stream>>>(kvf, kvbf, 65536, 256, 256);

    // ---- 6. depthwise conv, then tmp = (ql@kv)*zfac + fm, then osp = tmp@Wf+bf ----
    dwconv_kernel<<<NPATH, 256, 0, stream>>>(qkvf, dwc_w, dwc_b, fmbuf);
    gemm_f16<EPI_ROWSCALE_ADDAUX, false><<<dim3(2, 512, 1), 256, 0, stream>>>(
        qlbf, kvbf, fmbuf, NPATH, 256, 256, 256, 256, 256, nullptr, fmbuf, zfac, 0);
    cvt_f16_kernel<<<4096, 256, 0, stream>>>(fmbuf, tmpbf, (long long)NPATH * D_, D_, D_);
    gemm_f16<EPI_BIAS, false><<<dim3(2, 512, 1), 256, 0, stream>>>(
        tmpbf, wfbf, fmbuf, NPATH, 256, 256, 256, 256, 256, bfv, nullptr, nullptr, 0);
    float* osp = fmbuf;

    // ---- 7. gated scoring (big streams via WMMA, small scalar) ----
    cvt_f16_kernel<<<4096, 256, 0, stream>>>(ocp, ocpbf, (long long)NPATH * D_, D_, D_);
    gemm_f16<EPI_BIAS_TANH, false><<<dim3(2, 512, 1), 256, 0, stream>>>(
        ocpbf, wabf, gate, NPATH, 256, 256, 256, 256, 256, ba, nullptr, nullptr, 0);
    gemm_f16<EPI_BIAS_SIGMUL, false><<<dim3(2, 512, 1), 256, 0, stream>>>(
        ocpbf, wbbf, gate, NPATH, 256, 256, 256, 256, 256, bb, gate, nullptr, 0);
    wc_reduce<<<NPATH / 8, 256, 0, stream>>>(gate, Wc, bc, Acp);

    cvt_f16_kernel<<<4096, 256, 0, stream>>>(osp, ospbf, (long long)NPATH * D_, D_, D_);
    gemm_f16<EPI_BIAS_TANH, false><<<dim3(2, 512, 1), 256, 0, stream>>>(
        ospbf, wabf, gate, NPATH, 256, 256, 256, 256, 256, ba, nullptr, nullptr, 0);
    gemm_f16<EPI_BIAS_SIGMUL, false><<<dim3(2, 512, 1), 256, 0, stream>>>(
        ospbf, wbbf, gate, NPATH, 256, 256, 256, 256, 256, bb, gate, nullptr, 0);
    wc_reduce<<<NPATH / 8, 256, 0, stream>>>(gate, Wc, bc, Asp);

    gated_small<<<1, 256, 0, stream>>>(occ, Wa, ba, Wb, bb, Wc, bc, Acc);
    gated_small<<<1, 256, 0, stream>>>(osc, Wa, ba, Wb, bb, Wc, bc, Asc);

    // ---- 8. softmax pooling ----
    zero_kernel<<<4, 256, 0, stream>>>(hbuf, 1024);
    pool_stats<<<1, 256, 0, stream>>>(Acc, MC, stats + 0);
    pool_apply<<<1, 256, 0, stream>>>(Acc, occ, stats + 0, hcc, MC);
    pool_stats<<<1, 256, 0, stream>>>(Acp, NPATH, stats + 2);
    pool_apply<<<256, 256, 0, stream>>>(Acp, ocp, stats + 2, hcp, NPATH);
    pool_stats<<<1, 256, 0, stream>>>(Asc, MC, stats + 4);
    pool_apply<<<1, 256, 0, stream>>>(Asc, osc, stats + 4, hsc, MC);
    pool_stats<<<1, 256, 0, stream>>>(Asp, NPATH, stats + 6);
    pool_apply<<<256, 256, 0, stream>>>(Asp, osp, stats + 6, hsp, NPATH);

    // ---- 9. final fuse MLPs ----
    final_mlp<<<1, 256, 0, stream>>>(hcc, hcp, hsc, hsp,
                                     W1, b1, W2, b2, W3a, b3a, W3b, b3b, out);
}
